// AttnBlock_68642167325031
// MI455X (gfx1250) — matrix operand order repaired and checked by reading, not compiled, not run on hardware
//
#include <hip/hip_runtime.h>

typedef __attribute__((ext_vector_type(16))) _Float16 v16h;
typedef __attribute__((ext_vector_type(8)))  _Float16 v8h;
typedef __attribute__((ext_vector_type(4)))  _Float16 v4h;
typedef __attribute__((ext_vector_type(2)))  _Float16 v2h;
typedef __attribute__((ext_vector_type(8)))  float    v8f;

namespace {
constexpr int kS   = 2048;   // sequence length
constexpr int kD   = 64;     // head dim
constexpr int kBH  = 64;     // B*H
constexpr int BM   = 64;     // query rows per workgroup (16 per wave)
constexpr int BN   = 128;    // key columns per LDS tile
constexpr int NW   = 4;      // waves per workgroup
constexpr int KSTR = 72;     // Ks row stride (f16): 144B = 9*16B, conflict-free
constexpr int VSTR = 136;    // Vt/Ps row stride (f16): 272B = 17*16B, conflict-free
// scores computed directly in log2 domain: fold log2(e)/sqrt(D) into Q
constexpr float kQScale = 0.125f * 1.4426950408889634f;
}

// v_cvt_pk_f16_f32 wrapper (builtin returns __fp16 vector; bit-identical cast).
__device__ __forceinline__ v2h pk_f16(float a, float b) {
  return __builtin_bit_cast(v2h, __builtin_amdgcn_cvt_pkrtz(a, b));
}

// Build a 16-half fragment from two 16B LDS chunks.
__device__ __forceinline__ v16h ld_frag16(const _Float16* p0, const _Float16* p1) {
  v8h a = *(const v8h*)p0;
  v8h b = *(const v8h*)p1;
  v16h r;
#pragma unroll
  for (int i = 0; i < 8; ++i) { r[i] = a[i]; r[i + 8] = b[i]; }
  return r;
}

// XOR-butterfly via ds_swizzle_b32 (group-of-32 mode: imm = xor<<10 | 0x1f).
template <int IMM>
__device__ __forceinline__ float swz(float x) {
  return __int_as_float(__builtin_amdgcn_ds_swizzle(__float_as_int(x), IMM));
}

__device__ __forceinline__ v8f wmma_f16(v16h a, v16h b, v8f c) {
  return __builtin_amdgcn_wmma_f32_16x16x32_f16(false, a, false, b, (short)0, c,
                                                false, false);
}

// Raw v_exp_f32 (args are <= 0 and well inside normal range; masked -1e30 -> 0).
__device__ __forceinline__ float fast_exp2(float x) {
  return __builtin_amdgcn_exp2f(x);
}

extern "C" __global__ __launch_bounds__(NW * 32)
void fa2_fwd_wmma(const float* __restrict__ Q, const float* __restrict__ K,
                  const float* __restrict__ V, float* __restrict__ O) {
  __shared__ __align__(16) _Float16 Ks[BN * KSTR];          // K tile [key][d]
  __shared__ __align__(16) _Float16 Vt[(kD + 16) * VSTR];   // V^T tile [d][key]; row 64 = ones
  __shared__ __align__(16) _Float16 Ps[NW * 16 * VSTR];     // per-wave P bounce (C->A layout)

  const int tid  = threadIdx.x;
  const int lane = tid & 31;
  const int wave = tid >> 5;
  const int hi   = (lane >> 4) & 1;
  const int l16  = lane & 15;

  const int bh = blockIdx.y;
  const int qb = blockIdx.x * BM;
  const int q0 = qb + wave * 16;

  const float* __restrict__ Qb = Q + (size_t)bh * kS * kD;
  const float* __restrict__ Kb = K + (size_t)bh * kS * kD;
  const float* __restrict__ Vb = V + (size_t)bh * kS * kD;
  float*       __restrict__ Ob = O + (size_t)bh * kS * kD;

  // ---- ones row (d=64) / zero rows (65..79) feed the l-accumulating 5th subtile ----
  for (int i = tid; i < 16 * VSTR; i += NW * 32)
    Vt[kD * VSTR + i] = (i < VSTR) ? (_Float16)1.0f : (_Float16)0.0f;

  // ---- Q A-fragments, scaled by log2(e)/sqrt(D) (scores in log2 domain) ----
  v16h qa[2];
  {
    const float* qrow = Qb + (size_t)(q0 + l16) * kD;
#pragma unroll
    for (int kk = 0; kk < 2; ++kk) {
      const int d0 = kk * 32 + hi * 16;
      const int d1 = kk * 32 + hi * 16 + 8;
#pragma unroll
      for (int e = 0; e < 8; ++e) {
        qa[kk][e]     = (_Float16)(qrow[d0 + e] * kQScale);
        qa[kk][e + 8] = (_Float16)(qrow[d1 + e] * kQScale);
      }
    }
  }

  v8f oacc[5] = {};                  // [0..3]: O (16x64); [4]: col 64 = running l
  float mrow[8];
#pragma unroll
  for (int r = 0; r < 8; ++r) mrow[r] = -1e30f;

  const int jmax = qb / BN;          // causal: only tiles touching/below the diagonal
  for (int j = 0; j <= jmax; ++j) {
    const int kb0 = j * BN;
    const float* __restrict__ ks = Kb + (size_t)kb0 * kD;
    const float* __restrict__ vs = Vb + (size_t)kb0 * kD;

    if (j < jmax) {  // prefetch next tile (global_prefetch_b8), 256B/thread
      __builtin_prefetch(ks + BN * kD + tid * 64, 0, 1);
      __builtin_prefetch(vs + BN * kD + tid * 64, 0, 1);
    }

    // ---- staging: K -> Ks (pkrtz + b64 store), V -> Vt (pkrtz, transposed) ----
#pragma unroll 4
    for (int it = 0; it < 16; ++it) {
      const int elem = (it * 128 + tid) * 4;   // 128x64 floats, float4 per thread-iter
      const int kr = elem >> 6;
      const int d  = elem & 63;                // multiple of 4
      const float4 kv = *(const float4*)(ks + elem);
      const v2h klo = pk_f16(kv.x, kv.y);
      const v2h khi = pk_f16(kv.z, kv.w);
      *(v4h*)&Ks[kr * KSTR + d] = __builtin_shufflevector(klo, khi, 0, 1, 2, 3);
      const float4 vv = *(const float4*)(vs + elem);
      const v2h vlo = pk_f16(vv.x, vv.y);
      const v2h vhi = pk_f16(vv.z, vv.w);
      Vt[(d)     * VSTR + kr] = vlo[0];
      Vt[(d + 1) * VSTR + kr] = vlo[1];        // d16_hi store
      Vt[(d + 2) * VSTR + kr] = vhi[0];
      Vt[(d + 3) * VSTR + kr] = vhi[1];        // d16_hi store
    }
    __syncthreads();

    // ---- S = Q*K^T (16x128 per wave, log2 domain); straight-line ----
    v8f sacc[8];
#pragma unroll
    for (int ns = 0; ns < 8; ++ns) {
      const _Float16* kp = &Ks[(ns * 16 + l16) * KSTR + hi * 16];
      v16h b0 = ld_frag16(kp,      kp + 8);
      v16h b1 = ld_frag16(kp + 32, kp + 40);
      v8f c = {};
      c = wmma_f16(qa[0], b0, c);
      c = wmma_f16(qa[1], b1, c);
      sacc[ns] = c;
    }

    // ---- causal mask on the diagonal tile only (future keys -> -1e30 -> P=0) ----
    if (j == jmax) {
#pragma unroll
      for (int ns = 0; ns < 8; ++ns) {
        const int n = kb0 + ns * 16 + l16;   // C layout: n = lane%16
#pragma unroll
        for (int r = 0; r < 8; ++r)
          if (n > q0 + r + hi * 8) sacc[ns][r] = -1e30f;
      }
    }

    // ---- online softmax: row max via ds_swizzle butterfly ----
    v8f mv, av;
#pragma unroll
    for (int r = 0; r < 8; ++r) {
      float mx = sacc[0][r];
#pragma unroll
      for (int ns = 1; ns < 8; ++ns) mx = fmaxf(mx, sacc[ns][r]);  // -> v_max3_f32
      mx = fmaxf(mx, swz<0x041F>(mx));   // xor 1
      mx = fmaxf(mx, swz<0x081F>(mx));   // xor 2
      mx = fmaxf(mx, swz<0x101F>(mx));   // xor 4
      mx = fmaxf(mx, swz<0x201F>(mx));   // xor 8
      const float mnew = fmaxf(mrow[r], mx);
      av[r] = fast_exp2(mrow[r] - mnew); // alpha, raw v_exp_f32
      mrow[r] = mnew;
      mv[r] = mnew;
    }

    // P = exp2(S - m); row sums come for free from the ones-column WMMA below
#pragma unroll
    for (int ns = 0; ns < 8; ++ns) {
      v8f t = sacc[ns] - mv;             // vector sub -> pk/dual-issue
#pragma unroll
      for (int r = 0; r < 8; ++r)
        sacc[ns][r] = fast_exp2(t[r]);   // raw v_exp_f32 (TRANS)
    }

#pragma unroll
    for (int nd = 0; nd < 5; ++nd)
      oacc[nd] *= av;                    // v_pk_mul_f32

    // ---- P: C-layout regs -> per-wave LDS bounce (pkrtz over row pairs) ----
    _Float16* myP = &Ps[wave * 16 * VSTR];
#pragma unroll
    for (int ns = 0; ns < 8; ++ns) {
      const int c0 = ns * 16 + l16;
#pragma unroll
      for (int r = 0; r < 8; r += 2) {
        const v2h ph = pk_f16(sacc[ns][r], sacc[ns][r + 1]);
        myP[(r + 0 + hi * 8) * VSTR + c0] = ph[0];
        myP[(r + 1 + hi * 8) * VSTR + c0] = ph[1];   // d16_hi store
      }
    }

    // ---- O += P*V (+ l via ones row); unconditional 4 kk-chunks ----
#pragma unroll
    for (int kk = 0; kk < 4; ++kk) {
      const _Float16* prow = &myP[l16 * VSTR + kk * 32];
      v16h pa = ld_frag16(prow + hi * 16, prow + hi * 16 + 8);
#pragma unroll
      for (int nd = 0; nd < 5; ++nd) {
        const _Float16* vp = &Vt[(nd * 16 + l16) * VSTR + kk * 32 + hi * 16];
        v16h b = ld_frag16(vp, vp + 8);
        oacc[nd] = wmma_f16(pa, b, oacc[nd]);
      }
    }
    __syncthreads();   // protect Ks/Vt before next staging
  }

  // ---- epilogue: l lives in oacc[4] column n==64 (lane l16==0 of each half) ----
#pragma unroll
  for (int r = 0; r < 8; ++r) {
    const float l = __shfl(oacc[4][r], hi * 16, 32);
    const float invl = 1.0f / l;
    const int m = q0 + r + hi * 8;
#pragma unroll
    for (int nd = 0; nd < 4; ++nd)
      Ob[(size_t)m * kD + nd * 16 + l16] = oacc[nd][r] * invl;
  }
}

extern "C" void kernel_launch(void* const* d_in, const int* in_sizes, int n_in,
                              void* d_out, int out_size, void* d_ws, size_t ws_size,
                              hipStream_t stream) {
  (void)in_sizes; (void)n_in; (void)out_size; (void)d_ws; (void)ws_size;
  const float* q = (const float*)d_in[0];
  const float* k = (const float*)d_in[1];
  const float* v = (const float*)d_in[2];
  // d_in[3] is the causal mask; causality is applied analytically in-kernel.
  float* out = (float*)d_out;
  dim3 grid(kS / BM, kBH);
  dim3 block(NW * 32);
  fa2_fwd_wmma<<<grid, block, 0, stream>>>(q, k, v, out);
}
